// AttentionGlobalContextExpertFusion_49469433315517
// MI455X (gfx1250) — matrix operand order repaired and checked by reading, not compiled, not run on hardware
//
#include <hip/hip_runtime.h>

// MoE top-2 expert fusion for MI455X (gfx1250, wave32).
// out[b] = x[b] @ (w0*W_e0 + w1*W_e1) + (w0*b_e0 + w1*b_e1), experts combined
// in f32 at tile-load time so only ONE bf16-WMMA GEMM per batch is needed.

typedef float  v8f   __attribute__((ext_vector_type(8)));
typedef __bf16 v16bf __attribute__((ext_vector_type(16)));

namespace {

constexpr int B = 8, S = 2048, D = 1024, E = 8;
constexpr int BM = 128, BN = 128, BK = 32;
constexpr int THREADS = 256;   // 8 waves of 32

union FragU { v16bf v; uint4 q[2]; };

} // namespace

// ---------------- gating: top-2 + renormalized weights + expert counts -----
__global__ void moe_gate_kernel(const float* __restrict__ rs,
                                int* __restrict__ gidx, float* __restrict__ gw,
                                float* __restrict__ cnt) {
  const int t = threadIdx.x;
  if (t < E) cnt[t] = 0.0f;
  __syncthreads();
  if (t < B) {
    float best = -3.4e38f; int bi = 0;
    for (int e = 0; e < E; ++e) {
      float s = rs[t * E + e];
      if (s > best) { best = s; bi = e; }
    }
    float sec = -3.4e38f; int si = 0;
    for (int e = 0; e < E; ++e) {
      if (e == bi) continue;
      float s = rs[t * E + e];
      if (s > sec) { sec = s; si = e; }
    }
    const float inv = 1.0f / (best + sec + 1e-8f);
    gidx[t * 2 + 0] = bi;        gidx[t * 2 + 1] = si;
    gw[t * 2 + 0]   = best * inv; gw[t * 2 + 1]  = sec * inv;
    atomicAdd(&cnt[bi], 1.0f);
    atomicAdd(&cnt[si], 1.0f);
  }
}

// ---------------- fused expert GEMM via v_wmma_f32_16x16x32_bf16 -----------
__global__ __launch_bounds__(THREADS, 1)
void moe_wmma_kernel(const float* __restrict__ x,
                     const float* __restrict__ Wt,   // (E, D, D) row-major [d][f]
                     const float* __restrict__ Eb,   // (E, D)
                     const int* __restrict__ gidx,
                     const float* __restrict__ gw,
                     float* __restrict__ out) {
  __shared__ __bf16 Alds[BM * BK];  // x tile, [row s][k]
  __shared__ __bf16 Blds[BN * BK];  // combined-W tile, transposed [col f][k]

  const int b  = blockIdx.z;
  const int s0 = blockIdx.x * BM;
  const int f0 = blockIdx.y * BN;

  const int   e0 = gidx[b * 2 + 0], e1 = gidx[b * 2 + 1];
  const float w0 = gw[b * 2 + 0],   w1 = gw[b * 2 + 1];

  const float* xb = x  + (size_t)b  * S * D;
  const float* W0 = Wt + (size_t)e0 * D * D;
  const float* W1 = Wt + (size_t)e1 * D * D;

  const int tid  = threadIdx.x;
  const int lane = tid & 31;
  const int wave = tid >> 5;
  const int mOff = (wave >> 1) * 32;   // 4 wave-rows  -> 128 M
  const int nOff = (wave & 1) * 64;    // 2 wave-cols  -> 128 N
  const int half = lane >> 4;          // lane-half for WMMA K striping
  const int lrow = lane & 15;

  v8f acc[2][4];
  #pragma unroll
  for (int mi = 0; mi < 2; ++mi)
    #pragma unroll
    for (int ni = 0; ni < 4; ++ni)
      #pragma unroll
      for (int r = 0; r < 8; ++r) acc[mi][ni][r] = 0.0f;

  for (int ks = 0; ks < D; ks += BK) {
    // ---- stage A: 128x32 f32 of x -> bf16 LDS (row-major); hardware cvt ----
    {
      const float4* src = (const float4*)(xb + (size_t)s0 * D + ks);
      #pragma unroll
      for (int i = 0; i < 4; ++i) {
        int idx = tid + i * THREADS;        // 0..1023 float4s
        int row = idx >> 3;                 // 8 float4 per 32-wide row
        int c4  = idx & 7;
        float4 v = src[(size_t)row * (D / 4) + c4];
        __bf16* dst = &Alds[row * BK + c4 * 4];
        dst[0] = (__bf16)v.x; dst[1] = (__bf16)v.y;
        dst[2] = (__bf16)v.z; dst[3] = (__bf16)v.w;
      }
    }
    // ---- stage B: 32x128 f32 of (w0*W_e0 + w1*W_e1) -> bf16 LDS, transposed [f][k]
    {
      const float4* p0 = (const float4*)(W0 + (size_t)ks * D + f0);
      const float4* p1 = (const float4*)(W1 + (size_t)ks * D + f0);
      #pragma unroll
      for (int i = 0; i < 4; ++i) {
        int idx = tid + i * THREADS;        // 0..1023 float4s
        int d   = idx >> 5;                 // 32 float4 per 128-wide row
        int c4  = idx & 31;
        float4 a = p0[(size_t)d * (D / 4) + c4];
        float4 c = p1[(size_t)d * (D / 4) + c4];
        int f = c4 * 4;
        Blds[(f + 0) * BK + d] = (__bf16)(w0 * a.x + w1 * c.x);
        Blds[(f + 1) * BK + d] = (__bf16)(w0 * a.y + w1 * c.y);
        Blds[(f + 2) * BK + d] = (__bf16)(w0 * a.z + w1 * c.z);
        Blds[(f + 3) * BK + d] = (__bf16)(w0 * a.w + w1 * c.w);
      }
    }
    // prefetch next k-tile into cache (global_prefetch_b8)
    if (ks + BK < D) {
      __builtin_prefetch(xb + (size_t)(s0 + (tid & 127)) * D + ks + BK, 0, 1);
      __builtin_prefetch(W0 + (size_t)(ks + BK + (tid & 31)) * D + f0, 0, 1);
      __builtin_prefetch(W1 + (size_t)(ks + BK + (tid & 31)) * D + f0, 0, 1);
    }
    __syncthreads();

    // ---- load WMMA fragments from LDS ----
    // A (16x32 bf16): lane-half 0 holds K 0-7,16-23; half 1 holds K 8-15,24-31.
    v16bf af[2], bfrag[4];
    #pragma unroll
    for (int mi = 0; mi < 2; ++mi) {
      const __bf16* p = &Alds[(mOff + mi * 16 + lrow) * BK];
      FragU u;
      u.q[0] = *(const uint4*)(p + half * 8);        // K = half*8 + 0..7
      u.q[1] = *(const uint4*)(p + half * 8 + 16);   // K = half*8 + 16..23
      af[mi] = u.v;
    }
    // B (32x16 bf16): lane-half 0 holds K 0-7,16-23; half 1 holds K 8-15,24-31.
    #pragma unroll
    for (int ni = 0; ni < 4; ++ni) {
      const __bf16* p = &Blds[(nOff + ni * 16 + lrow) * BK + half * 8];
      FragU u;
      u.q[0] = *(const uint4*)(p);
      u.q[1] = *(const uint4*)(p + 16);
      bfrag[ni] = u.v;
    }

    // ---- 8 WMMAs per wave per k-step ----
    #pragma unroll
    for (int mi = 0; mi < 2; ++mi)
      #pragma unroll
      for (int ni = 0; ni < 4; ++ni)
        acc[mi][ni] = __builtin_amdgcn_wmma_f32_16x16x32_bf16(
            false, af[mi], false, bfrag[ni], (short)0, acc[mi][ni],
            false, false);

    __syncthreads();
  }

  // ---- epilogue: combined bias + store f32 (C/D layout: VGPR r -> M=r+8*half)
  #pragma unroll
  for (int ni = 0; ni < 4; ++ni) {
    const int ncol = f0 + nOff + ni * 16 + lrow;
    const float bias = w0 * Eb[(size_t)e0 * D + ncol]
                     + w1 * Eb[(size_t)e1 * D + ncol];
    #pragma unroll
    for (int mi = 0; mi < 2; ++mi) {
      #pragma unroll
      for (int r = 0; r < 8; ++r) {
        const int mrow = s0 + mOff + mi * 16 + r + 8 * half;
        out[(size_t)b * S * D + (size_t)mrow * D + ncol] = acc[mi][ni][r] + bias;
      }
    }
  }
}

extern "C" void kernel_launch(void* const* d_in, const int* in_sizes, int n_in,
                              void* d_out, int out_size, void* d_ws, size_t ws_size,
                              hipStream_t stream) {
  (void)in_sizes; (void)n_in; (void)out_size; (void)ws_size;
  const float* x  = (const float*)d_in[0];   // (B,S,D)
  const float* rs = (const float*)d_in[1];   // (B,E)
  const float* Wt = (const float*)d_in[2];   // (E,D,D)
  const float* Eb = (const float*)d_in[3];   // (E,D)
  // d_in[4] is k (==2 in the reference setup); this kernel implements top-2.

  int*   gidx = (int*)d_ws;                                  // 2*B ints
  float* gw   = (float*)((char*)d_ws + 2 * B * sizeof(int)); // 2*B floats
  float* out  = (float*)d_out;                               // (B,S,D) f32
  float* cnt  = out + (size_t)B * S * D;                     // (E,) counts tail

  hipLaunchKernelGGL(moe_gate_kernel, dim3(1), dim3(64), 0, stream,
                     rs, gidx, gw, cnt);

  dim3 grid(S / BM, D / BN, B);   // 16 x 8 x 8 = 1024 blocks
  hipLaunchKernelGGL(moe_wmma_kernel, grid, dim3(THREADS), 0, stream,
                     x, Wt, Eb, gidx, gw, out);
}
